// FlashMHA_37366215475915
// MI455X (gfx1250) — matrix operand order repaired and checked by reading, not compiled, not run on hardware
//
#include <hip/hip_runtime.h>
#include <stdint.h>

// ---------- types ----------
typedef __attribute__((ext_vector_type(16))) __bf16 v16bf;
typedef __attribute__((ext_vector_type(8)))  float  v8f;
typedef __attribute__((ext_vector_type(4)))  unsigned int uv4;
typedef __attribute__((ext_vector_type(4)))  float  fv4;

union Frag {
    v16bf v;
    unsigned short u[16];
    uv4 q[2];
};

__device__ __forceinline__ unsigned short f2bf(float f) {
    unsigned int u = __builtin_bit_cast(unsigned int, f);
    u += 0x7FFFu + ((u >> 16) & 1u);           // round-to-nearest-even
    return (unsigned short)(u >> 16);
}

__device__ __forceinline__ v8f wmma_bf16(const Frag& a, const Frag& b, v8f c) {
    // (neg_a, A, neg_b, B, c_mod, C, reuse_a, reuse_b)
    return __builtin_amdgcn_wmma_f32_16x16x32_bf16(false, a.v, false, b.v,
                                                   (short)0, c, false, false);
}

// Async copy 32 bytes global -> LDS (per lane). CDNA5 GLOBAL_LOAD_ASYNC_TO_LDS,
// tracked by ASYNCcnt. INST_OFFSET is added to BOTH the LDS and global address
// (ISA 08_async_tensor §4.4), so one addr pair covers both 16B halves.
// LDS offset = low 32 bits of the generic pointer (aperture rule).
__device__ __forceinline__ void async_copy_b32g(const void* gsrc, void* ldst) {
    unsigned lds = (unsigned)(uintptr_t)ldst;
    asm volatile(
        "global_load_async_to_lds_b128 %0, %1, off\n\t"
        "global_load_async_to_lds_b128 %0, %1, off offset:16"
        :: "v"(lds), "v"(gsrc) : "memory");
}
__device__ __forceinline__ void async_wait0() {
    asm volatile("s_wait_asynccnt 0" ::: "memory");
}

// =====================================================================
// Generic GEMM:  C[M,N] = A[M,K] @ W[N,K]^T + bias[N]
// BM=128, BN=128, BK=32, 256 threads = 8 waves, wave grid 4 (M) x 2 (N)
// A_F32: A is fp32 (convert to bf16 while staging) else A is bf16
//        (staged via GLOBAL_LOAD_ASYNC_TO_LDS).
// OUT_BF16: write bf16, else fp32.
// =====================================================================
template<bool A_F32, bool OUT_BF16>
__global__ __launch_bounds__(256) void gemm_wmma(
    const void* __restrict__ Av, const float* __restrict__ W,
    const float* __restrict__ bias, void* __restrict__ Cv,
    int M, int N, int K) {
    __shared__ unsigned short As[128 * 40];   // 32 K-halves + 8 pad (80B rows, 16B aligned)
    __shared__ unsigned short Ws[128 * 40];

    const int tid  = threadIdx.x;
    const int lane = tid & 31;
    const int wv   = tid >> 5;
    const int l15  = lane & 15;
    const int kh   = lane >> 4;               // half-wave selector
    const int wm   = (wv & 3) * 32;           // wave row base in tile
    const int wn   = (wv >> 2) * 64;          // wave col base in tile
    const int mBase = blockIdx.y * 128;
    const int nBase = blockIdx.x * 128;

    const int srow = tid >> 1;                // staging row 0..127
    const int kseg = (tid & 1) * 16;          // staging K segment

    v8f acc[8];
#pragma unroll
    for (int i = 0; i < 8; ++i) acc[i] = (v8f){};

    for (int kt = 0; kt < K; kt += 32) {
        __syncthreads();
        // ---- stage A tile (bf16) ----
        {
            unsigned short* dst = &As[srow * 40 + kseg];
            if (A_F32) {
                const float* ap = (const float*)Av + (size_t)(mBase + srow) * K + kt + kseg;
                uv4 pk0 = (uv4){}, pk1 = (uv4){};
#pragma unroll
                for (int i = 0; i < 4; ++i) {
                    fv4 f = *(const fv4*)(ap + i * 4);
                    unsigned int lo = ((unsigned int)f2bf(f.y) << 16) | f2bf(f.x);
                    unsigned int hi = ((unsigned int)f2bf(f.w) << 16) | f2bf(f.z);
                    if (i < 2) { pk0[i * 2] = lo; pk0[i * 2 + 1] = hi; }
                    else       { pk1[(i - 2) * 2] = lo; pk1[(i - 2) * 2 + 1] = hi; }
                }
                *(uv4*)dst = pk0; *(uv4*)(dst + 8) = pk1;
            } else {
                const unsigned short* ap =
                    (const unsigned short*)Av + (size_t)(mBase + srow) * K + kt + kseg;
                async_copy_b32g(ap, dst);      // no conversion -> async DMA path
            }
        }
        // ---- stage W tile (fp32 -> bf16, VALU convert) ----
        {
            const float* wp = W + (size_t)(nBase + srow) * K + kt + kseg;
            unsigned short* dst = &Ws[srow * 40 + kseg];
            uv4 pk0 = (uv4){}, pk1 = (uv4){};
#pragma unroll
            for (int i = 0; i < 4; ++i) {
                fv4 f = *(const fv4*)(wp + i * 4);
                unsigned int lo = ((unsigned int)f2bf(f.y) << 16) | f2bf(f.x);
                unsigned int hi = ((unsigned int)f2bf(f.w) << 16) | f2bf(f.z);
                if (i < 2) { pk0[i * 2] = lo; pk0[i * 2 + 1] = hi; }
                else       { pk1[(i - 2) * 2] = lo; pk1[(i - 2) * 2 + 1] = hi; }
            }
            *(uv4*)dst = pk0; *(uv4*)(dst + 8) = pk1;
        }
        // prefetch next K-step staging lines into cache while we compute
        if (kt + 32 < K) {
            __builtin_prefetch((const float*)W + (size_t)(nBase + srow) * K + kt + 32 + kseg, 0, 0);
            if (A_F32)
                __builtin_prefetch((const float*)Av + (size_t)(mBase + srow) * K + kt + 32 + kseg, 0, 0);
        }
        if (!A_F32) async_wait0();             // ASYNCcnt not covered by barrier
        __syncthreads();

        // ---- compute: 2 M-tiles x 4 N-tiles per wave ----
        Frag a0, a1;
        {
            // A 16x32 bf16 layout: lane(row)=l15, halves: K = kh*8+[0..7], 16+kh*8+[0..7]
            const unsigned short* ap0 = &As[(wm + l15) * 40];
            a0.q[0] = *(const uv4*)(ap0 + kh * 8);
            a0.q[1] = *(const uv4*)(ap0 + 16 + kh * 8);
            const unsigned short* ap1 = &As[(wm + 16 + l15) * 40];
            a1.q[0] = *(const uv4*)(ap1 + kh * 8);
            a1.q[1] = *(const uv4*)(ap1 + 16 + kh * 8);
        }
#pragma unroll
        for (int j = 0; j < 4; ++j) {
            Frag bfr;
            // B 32x16 layout: lane(col)=l15, K = kh*8+[0..7], 16+kh*8+[0..7]
            const unsigned short* bp = &Ws[(wn + j * 16 + l15) * 40 + kh * 8];
            bfr.q[0] = ((const uv4*)bp)[0];
            bfr.q[1] = *(const uv4*)(bp + 16);
            acc[j]     = wmma_bf16(a0, bfr, acc[j]);
            acc[4 + j] = wmma_bf16(a1, bfr, acc[4 + j]);
        }
    }

    // ---- epilogue: C layout VGPR r -> (row = base + r + 8*kh, col = lane&15) ----
#pragma unroll
    for (int mt = 0; mt < 2; ++mt) {
#pragma unroll
        for (int j = 0; j < 4; ++j) {
            v8f c = acc[mt * 4 + j];
            int col  = nBase + wn + j * 16 + l15;
            int row0 = mBase + wm + mt * 16 + kh * 8;
            float bv = bias[col];
#pragma unroll
            for (int r = 0; r < 8; ++r) {
                float val = c[r] + bv;
                size_t off = (size_t)(row0 + r) * N + col;
                if (OUT_BF16) ((unsigned short*)Cv)[off] = f2bf(val);
                else          ((float*)Cv)[off] = val;
            }
        }
    }
}

// =====================================================================
// Attention: one workgroup per (batch, 32-row query tile), loops 16 heads.
// Scores kept f32 in LDS (32 x 1024, ~129KB -> exploits 320KB WGP LDS).
// Q/V tiles staged with GLOBAL_LOAD_ASYNC_TO_LDS (pure bf16 copies).
// =====================================================================
#define ATT_S   1024
#define ATT_D   2048
#define SPITCH  1032   // score row pitch (floats), 16B-aligned rows
#define QPITCH  136    // Q/V row pitch (halves), 16B-aligned rows
#define ATT_SMEM (32 * SPITCH * 4 + 2 * 32 * QPITCH * 2 + 32 * 8 * 4)

__global__ __launch_bounds__(256) void attn_kernel(
    const unsigned short* __restrict__ q2,
    const unsigned short* __restrict__ k2,
    const unsigned short* __restrict__ v2,
    unsigned short* __restrict__ ctx,
    float* __restrict__ attn_out) {
    extern __shared__ char smem[];
    float*          sS   = (float*)smem;                                   // 32 x SPITCH f32
    unsigned short* sQ   = (unsigned short*)(smem + 32 * SPITCH * 4);      // 32 x QPITCH bf16
    unsigned short* sV   = sQ + 32 * QPITCH;                               // 32 x QPITCH bf16
    float*          sRed = (float*)(sV + 32 * QPITCH);                     // 32 x 8 f32

    const int tid  = threadIdx.x;
    const int lane = tid & 31;
    const int wv   = tid >> 5;
    const int l15  = lane & 15;
    const int kh   = lane >> 4;
    const int b    = blockIdx.y;
    const int t0   = blockIdx.x * 32;
    const float scale = 0.08838834764831845f;   // 1/sqrt(128)
    const float invH  = 1.0f / 16.0f;

    const int srow = tid >> 3;                  // staging row 0..31
    const int dseg = (tid & 7) * 16;            // staging halves segment

    for (int h = 0; h < 16; ++h) {
        const size_t hoff = (size_t)h * 128;

        // ---- stage Q tile (async global->LDS) ----
        {
            const unsigned short* src =
                q2 + ((size_t)(b * ATT_S + t0 + srow) * ATT_D + hoff + dseg);
            async_copy_b32g(src, &sQ[srow * QPITCH + dseg]);
            async_wait0();
        }
        __syncthreads();

        // ---- scores = (Q K^T) * scale ; K fragments straight from global ----
        for (int j = 0; j < 8; ++j) {
            int nt = wv * 8 + j;                 // 64 column tiles / 8 waves
            v8f s0 = (v8f){}, s1 = (v8f){};
            const unsigned short* kp =
                k2 + ((size_t)(b * ATT_S + nt * 16 + l15) * ATT_D + hoff);
#pragma unroll
            for (int ks = 0; ks < 4; ++ks) {
                Frag a0, a1, bfr;
                const unsigned short* ap0 = &sQ[l15 * QPITCH + ks * 32];
                a0.q[0] = *(const uv4*)(ap0 + kh * 8);
                a0.q[1] = *(const uv4*)(ap0 + 16 + kh * 8);
                const unsigned short* ap1 = &sQ[(16 + l15) * QPITCH + ks * 32];
                a1.q[0] = *(const uv4*)(ap1 + kh * 8);
                a1.q[1] = *(const uv4*)(ap1 + 16 + kh * 8);
                const unsigned short* bp = kp + ks * 32 + kh * 8;
                bfr.q[0] = ((const uv4*)bp)[0];
                bfr.q[1] = *(const uv4*)(bp + 16);
                s0 = wmma_bf16(a0, bfr, s0);
                s1 = wmma_bf16(a1, bfr, s1);
            }
            int col = nt * 16 + l15;
#pragma unroll
            for (int r = 0; r < 8; ++r) {
                sS[(kh * 8 + r) * SPITCH + col]      = s0[r] * scale;
                sS[(16 + kh * 8 + r) * SPITCH + col] = s1[r] * scale;
            }
        }
        __syncthreads();

        // ---- softmax: 8 threads per row ----
        {
            const int r = tid >> 3, lr = tid & 7;
            float* rowp = &sS[r * SPITCH];
            float pmax = -3.4e38f;
            for (int i = 0; i < 128; ++i) pmax = fmaxf(pmax, rowp[lr + i * 8]);
            sRed[r * 8 + lr] = pmax;
            __syncthreads();
            float rowmax = sRed[r * 8];
#pragma unroll
            for (int i = 1; i < 8; ++i) rowmax = fmaxf(rowmax, sRed[r * 8 + i]);
            __syncthreads();
            float psum = 0.f;
            for (int i = 0; i < 128; ++i) {
                float e = __expf(rowp[lr + i * 8] - rowmax);
                rowp[lr + i * 8] = e;
                psum += e;
            }
            sRed[r * 8 + lr] = psum;
            __syncthreads();
            float rowsum = 0.f;
#pragma unroll
            for (int i = 0; i < 8; ++i) rowsum += sRed[r * 8 + i];
            float inv = 1.0f / rowsum;
            for (int i = 0; i < 128; ++i) rowp[lr + i * 8] *= inv;
        }
        __syncthreads();

        // ---- accumulate head-mean attention weights (workgroup owns slice) ----
        {
            const int tr = tid >> 3;
            const int c0 = (tid & 7) * 128;
            float* op = attn_out + ((size_t)(b * ATT_S + t0 + tr) * ATT_S + c0);
            const float* pp = &sS[tr * SPITCH + c0];
            for (int i = 0; i < 32; ++i) {
                fv4 p = ((const fv4*)pp)[i];
                fv4 o = p * invH;
                if (h != 0) o += ((const fv4*)op)[i];
                ((fv4*)op)[i] = o;
            }
        }

        // ---- ctx = P @ V ; wave w owns 16 head-dims, K-loop over S ----
        v8f c0v = (v8f){}, c1v = (v8f){};
        for (int ks = 0; ks < 32; ++ks) {
            __syncthreads();
            {
                const unsigned short* vsrc =
                    v2 + ((size_t)(b * ATT_S + ks * 32 + srow) * ATT_D + hoff + dseg);
                async_copy_b32g(vsrc, &sV[srow * QPITCH + dseg]);
                async_wait0();
            }
            __syncthreads();
            Frag a0, a1, bfr;
            const int kb = ks * 32;
#pragma unroll
            for (int e = 0; e < 8; ++e) {
                a0.u[e]     = f2bf(sS[l15 * SPITCH + kb + kh * 8 + e]);
                a0.u[e + 8] = f2bf(sS[l15 * SPITCH + kb + 16 + kh * 8 + e]);
                a1.u[e]     = f2bf(sS[(16 + l15) * SPITCH + kb + kh * 8 + e]);
                a1.u[e + 8] = f2bf(sS[(16 + l15) * SPITCH + kb + 16 + kh * 8 + e]);
            }
            const int colv = wv * 16 + l15;
#pragma unroll
            for (int e = 0; e < 16; ++e)
                bfr.u[e] = sV[(e < 8 ? kh * 8 + e : 16 + kh * 8 + (e - 8)) * QPITCH + colv];
            c0v = wmma_bf16(a0, bfr, c0v);
            c1v = wmma_bf16(a1, bfr, c1v);
        }
        // ---- write ctx tile (bf16, (B,S,D) with head offset) ----
        {
            int gcol = (int)hoff + wv * 16 + l15;
#pragma unroll
            for (int r = 0; r < 8; ++r) {
                int grow0 = t0 + kh * 8 + r;
                int grow1 = t0 + 16 + kh * 8 + r;
                ctx[(size_t)(b * ATT_S + grow0) * ATT_D + gcol] = f2bf(c0v[r]);
                ctx[(size_t)(b * ATT_S + grow1) * ATT_D + gcol] = f2bf(c1v[r]);
            }
        }
        __syncthreads();
    }
}

// =====================================================================
extern "C" void kernel_launch(void* const* d_in, const int* in_sizes, int n_in,
                              void* d_out, int out_size, void* d_ws, size_t ws_size,
                              hipStream_t stream) {
    (void)in_sizes; (void)n_in; (void)out_size; (void)ws_size;
    const int Bn = 4, S = 1024, D = 2048;
    const int M = Bn * S;                      // 4096
    const size_t MN = (size_t)M * D;           // 8,388,608 elements
    const size_t WD = (size_t)D * D;

    const float* q   = (const float*)d_in[0];
    const float* k   = (const float*)d_in[1];
    const float* v   = (const float*)d_in[2];
    const float* ipw = (const float*)d_in[3];
    const float* ipb = (const float*)d_in[4];
    const float* iiw = (const float*)d_in[5];
    const float* iib = (const float*)d_in[6];
    const float* iow = (const float*)d_in[7];
    const float* iob = (const float*)d_in[8];
    const float* ow  = (const float*)d_in[9];
    const float* ob  = (const float*)d_in[10];

    unsigned short* ws = (unsigned short*)d_ws;
    unsigned short* q1 = ws;
    unsigned short* k1 = ws + MN;
    unsigned short* v1 = ws + 2 * MN;
    unsigned short* q2 = ws + 3 * MN;
    unsigned short* k2 = ws + 4 * MN;
    unsigned short* v2 = ws + 5 * MN;
    unsigned short* ctx = q1;                  // reuse after inner projections
    unsigned short* io  = k1;                  // reuse

    float* out      = (float*)d_out;
    float* attn_out = out + MN;

    dim3 grid(D / 128, M / 128);               // (16, 32)
    dim3 blk(256);

    // outer packed in-projection
    gemm_wmma<true,  true ><<<grid, blk, 0, stream>>>(q, ipw,          ipb,          q1, M, D, D);
    gemm_wmma<true,  true ><<<grid, blk, 0, stream>>>(k, ipw + WD,     ipb + D,      k1, M, D, D);
    gemm_wmma<true,  true ><<<grid, blk, 0, stream>>>(v, ipw + 2 * WD, ipb + 2 * D,  v1, M, D, D);
    // inner packed in-projection
    gemm_wmma<false, true ><<<grid, blk, 0, stream>>>(q1, iiw,          iib,         q2, M, D, D);
    gemm_wmma<false, true ><<<grid, blk, 0, stream>>>(k1, iiw + WD,     iib + D,     k2, M, D, D);
    gemm_wmma<false, true ><<<grid, blk, 0, stream>>>(v1, iiw + 2 * WD, iib + 2 * D, v2, M, D, D);
    // attention (+ head-mean weights into d_out)
    hipFuncSetAttribute((const void*)attn_kernel,
                        hipFuncAttributeMaxDynamicSharedMemorySize, (int)ATT_SMEM);
    attn_kernel<<<dim3(S / 32, Bn), blk, ATT_SMEM, stream>>>(q2, k2, v2, ctx, attn_out);
    // inner out_proj then outer out_proj
    gemm_wmma<false, true ><<<grid, blk, 0, stream>>>(ctx, iow, iob, io,  M, D, D);
    gemm_wmma<false, false><<<grid, blk, 0, stream>>>(io,  ow,  ob,  out, M, D, D);
}
